// MHA_59201829208729
// MI455X (gfx1250) — matrix operand order repaired and checked by reading, not compiled, not run on hardware
//
#include <hip/hip_runtime.h>

typedef __bf16 bf16_t;
typedef __attribute__((ext_vector_type(16))) __bf16 v16bf;
typedef __attribute__((ext_vector_type(8)))  __bf16 v8bf;
typedef __attribute__((ext_vector_type(8)))  float  v8f;
typedef __attribute__((ext_vector_type(4)))  float  f32x4;
typedef __attribute__((ext_vector_type(4)))  unsigned int u32x4;

constexpr int BATCH = 2, SEQ = 2048, DIM = 2048, NH = 16, HD = 128;
constexpr int M_ROWS = BATCH * SEQ;            // 4096
constexpr int BM = 128, BN = 128, BK = 64;     // GEMM tiles
constexpr int BQ = 64, BKV = 64;               // attention tiles
constexpr float ATTN_SCALE = 0.08838834764831845f; // 1/sqrt(128)

// ---------------- WMMA fragment helpers (per CDNA5 ISA layouts) ----------------
// A 16x32 bf16: lane l: M = l%16; element j -> K = j + (l/16)*8 + (j>=8 ? 8 : 0)
__device__ __forceinline__ v16bf load_a_frag(const bf16_t* row, int half) {
  union { v16bf v; v8bf h[2]; } u;
  u.h[0] = *(const v8bf*)(row + half * 8);
  u.h[1] = *(const v8bf*)(row + 16 + half * 8);
  return u.v;
}
// B 32x16 bf16: lane l: N = l%16; element j -> K = j + (l/16)*8 + (j>=8 ? 8 : 0)
__device__ __forceinline__ v16bf load_b_frag(const bf16_t* row, int half) {
  union { v16bf v; v8bf h[2]; } u;
  u.h[0] = *(const v8bf*)(row + half * 8);
  u.h[1] = *(const v8bf*)(row + 16 + half * 8);
  return u.v;
}
__device__ __forceinline__ v8f wmma_bf16(v16bf a, v16bf b, v8f c) {
  return __builtin_amdgcn_wmma_f32_16x16x32_bf16(false, a, false, b, (short)0, c,
                                                 false, false);
}

// ---------------- CDNA5 async memory->LDS (ASYNCcnt) ---------------------------
// Generic pointers into __shared__ carry the wave-relative LDS byte offset in
// their low 32 bits (ISA 10.2: LDS aperture check is on addr[63:32] only).
__device__ __forceinline__ unsigned lds_addr(const void* p) {
  return (unsigned)(uintptr_t)p;
}
// GVS mode: global = SGPR64 base + per-lane signed i32 byte offset.
// Copies 16 bytes memory -> LDS per active lane, tracked by ASYNCcnt.
__device__ __forceinline__ void async_copy_b128(unsigned lds_byte,
                                                const bf16_t* base,
                                                int byte_off) {
  asm volatile("global_load_async_to_lds_b128 %0, %1, %2"
               :: "v"(lds_byte), "v"(byte_off), "s"(base)
               : "memory");
}
__device__ __forceinline__ void wait_asynccnt0() {
  asm volatile("s_wait_asynccnt 0x0" ::: "memory");
}

// ---------------- Tiled GEMM: C = A @ B + bias --------------------------------
// A_BF16: A is bf16 (ctx, staged via async-to-LDS) vs fp32 (x, converted in flight)
// QKV_OUT: scatter bf16 into q/k/v [B,H,S,HD]; else store fp32 row-major
template <bool A_BF16, bool QKV_OUT>
__global__ __launch_bounds__(256) void gemm_wmma_kernel(
    const float* __restrict__ Af, const bf16_t* __restrict__ Ab,
    const float* __restrict__ Bw, const float* __restrict__ bias,
    bf16_t* __restrict__ qb, bf16_t* __restrict__ kb, bf16_t* __restrict__ vb,
    float* __restrict__ outf, int N, int K) {
  __shared__ bf16_t As[BM][BK + 8];   // row-major [m][k]
  __shared__ bf16_t Bt[BN][BK + 8];   // transposed [n][k]

  const int tid = threadIdx.x;
  const int wave = tid >> 5, lane = tid & 31;
  const int half = lane >> 4, l16 = lane & 15;
  const int bm = blockIdx.y * BM, bn = blockIdx.x * BN;
  const int wm = (wave >> 1) * 32;    // 4 waves along M
  const int wn = (wave & 1) * 64;     // 2 waves along N

  v8f acc[2][4] = {};

  for (int k0 = 0; k0 < K; k0 += BK) {
    __syncthreads();
    if (A_BF16) {
      // A tile 128x64 bf16: pure async DMA, 4 x 16B chunks per thread
      const bf16_t* base = Ab + (size_t)bm * K + k0;
#pragma unroll
      for (int j = 0; j < 4; ++j) {
        const int e = (tid + j * 256) * 8;  // linear bf16 element in tile
        const int r = e >> 6, c = e & 63;
        async_copy_b128(lds_addr(&As[r][c]), base, 2 * (r * K + c));
      }
    } else {
      // A tile 128x64 fp32 -> bf16 convert in flight
      const int r = tid >> 1, c0 = (tid & 1) * 32;
      const float* src = Af + (size_t)(bm + r) * K + k0 + c0;
#pragma unroll
      for (int j = 0; j < 4; ++j) {
        union { u32x4 u; bf16_t e[8]; } pk;
        f32x4 f0 = *(const f32x4*)(src + j * 8);
        f32x4 f1 = *(const f32x4*)(src + j * 8 + 4);
        pk.e[0] = (bf16_t)f0.x; pk.e[1] = (bf16_t)f0.y;
        pk.e[2] = (bf16_t)f0.z; pk.e[3] = (bf16_t)f0.w;
        pk.e[4] = (bf16_t)f1.x; pk.e[5] = (bf16_t)f1.y;
        pk.e[6] = (bf16_t)f1.z; pk.e[7] = (bf16_t)f1.w;
        *(u32x4*)&As[r][c0 + j * 8] = pk.u;
      }
    }
    {  // B tile 64x128 fp32, store transposed bf16: 32 elems per thread
      const int kk = tid >> 2, n0 = (tid & 3) * 32;
      const float* src = Bw + (size_t)(k0 + kk) * N + bn + n0;
#pragma unroll
      for (int j = 0; j < 8; ++j) {
        f32x4 f = *(const f32x4*)(src + j * 4);
        Bt[n0 + j * 4 + 0][kk] = (bf16_t)f.x;
        Bt[n0 + j * 4 + 1][kk] = (bf16_t)f.y;
        Bt[n0 + j * 4 + 2][kk] = (bf16_t)f.z;
        Bt[n0 + j * 4 + 3][kk] = (bf16_t)f.w;
      }
    }
    if (A_BF16) wait_asynccnt0();
    __syncthreads();

#pragma unroll
    for (int ks = 0; ks < BK; ks += 32) {
      v16bf a0 = load_a_frag(&As[wm + l16][ks], half);
      v16bf a1 = load_a_frag(&As[wm + 16 + l16][ks], half);
#pragma unroll
      for (int j = 0; j < 4; ++j) {
        v16bf bfr = load_b_frag(&Bt[wn + j * 16 + l16][ks], half);
        acc[0][j] = wmma_bf16(a0, bfr, acc[0][j]);
        acc[1][j] = wmma_bf16(a1, bfr, acc[1][j]);
      }
    }
  }

  // Epilogue. C layout: lane -> N = l%16; vgpr r -> M = r + 8*(l/16)
#pragma unroll
  for (int i = 0; i < 2; ++i)
#pragma unroll
    for (int j = 0; j < 4; ++j) {
      const int colg = bn + wn + j * 16 + l16;
      const float bv = bias[colg];
#pragma unroll
      for (int r = 0; r < 8; ++r) {
        const int rowg = bm + wm + i * 16 + r + half * 8;
        const float v = acc[i][j][r] + bv;
        if (QKV_OUT) {
          const int which = colg >> 11;           // col / 2048
          const int h = (colg >> 7) & (NH - 1);   // head
          const int d = colg & (HD - 1);
          const int bidx = rowg >> 11;            // row / SEQ
          const int sidx = rowg & (SEQ - 1);
          bf16_t* dst = (which == 0) ? qb : (which == 1) ? kb : vb;
          dst[(((size_t)bidx * NH + h) * SEQ + sidx) * HD + d] = (bf16_t)v;
        } else {
          outf[(size_t)rowg * N + colg] = v;
        }
      }
    }
}

// ---------------- RoPE on q and k (fp32 math, bf16 storage) --------------------
__global__ __launch_bounds__(256) void rope_kernel(bf16_t* __restrict__ qb,
                                                   bf16_t* __restrict__ kb) {
  const int i = blockIdx.x * blockDim.x + threadIdx.x;
  const int d = i & 63;
  const int s = (i >> 6) & (SEQ - 1);
  const int bh = i >> 17;  // / (64*SEQ)
  if (bh >= BATCH * NH) return;
  const size_t base = ((size_t)bh * SEQ + s) * HD;
  const float inv = __expf(-0.14391156646f * (float)d);  // 10000^(-d/64)
  const float fr = (float)s * inv;
  const float c = cosf(fr), si = sinf(fr);
  {
    float t1 = (float)qb[base + d], t2 = (float)qb[base + d + 64];
    qb[base + d]      = (bf16_t)(t1 * c - t2 * si);
    qb[base + d + 64] = (bf16_t)(t1 * si + t2 * c);
  }
  {
    float t1 = (float)kb[base + d], t2 = (float)kb[base + d + 64];
    kb[base + d]      = (bf16_t)(t1 * c - t2 * si);
    kb[base + d + 64] = (bf16_t)(t1 * si + t2 * c);
  }
}

// ---------------- Causal flash attention (bf16 WMMA, fp32 softmax) -------------
__global__ __launch_bounds__(128) void flash_attn_kernel(
    const bf16_t* __restrict__ qb, const bf16_t* __restrict__ kb,
    const bf16_t* __restrict__ vb, bf16_t* __restrict__ ctx) {
  __shared__ bf16_t Qs[BQ][HD + 8];        // 17.4 KB
  __shared__ bf16_t Ks[BKV][HD + 8];       // 17.4 KB
  __shared__ bf16_t Vt[HD][BKV + 8];       // 18.4 KB (V transposed [d][key])
  __shared__ bf16_t Ps[4][16][BKV + 8];    //  9.2 KB (per-wave P scratch)

  const int tid = threadIdx.x;
  const int wave = tid >> 5, lane = tid & 31;
  const int half = lane >> 4, l16 = lane & 15;
  const int bh = blockIdx.x;
  const int q0 = blockIdx.y * BQ;
  const size_t base = (size_t)bh * SEQ * HD;

  {  // stage Q tile 64x128 via async DMA: 8 x 16B chunks per thread
    const bf16_t* src = qb + base + (size_t)q0 * HD;
#pragma unroll
    for (int j = 0; j < 8; ++j) {
      const int e = (tid + j * 128) * 8;   // linear bf16 element in tile
      const int r = e >> 7, c = e & 127;
      async_copy_b128(lds_addr(&Qs[r][c]), src, 2 * e);
    }
  }
  wait_asynccnt0();
  __syncthreads();

  v16bf qf[4];
#pragma unroll
  for (int ksp = 0; ksp < 4; ++ksp)
    qf[ksp] = load_a_frag(&Qs[wave * 16 + l16][ksp * 32], half);

  float mi[8], li[8];
  v8f oacc[8] = {};
#pragma unroll
  for (int r = 0; r < 8; ++r) { mi[r] = -1e30f; li[r] = 0.f; }

  const int myqmax = q0 + wave * 16 + 15;
  const int nkb = q0 / BKV + 1;  // causal: keys < q0 + BQ

  for (int kb0 = 0; kb0 < nkb; ++kb0) {
    __syncthreads();
    {  // stage K tile 64x128 via async DMA
      const bf16_t* src = kb + base + (size_t)(kb0 * BKV) * HD;
#pragma unroll
      for (int j = 0; j < 8; ++j) {
        const int e = (tid + j * 128) * 8;
        const int r = e >> 7, c = e & 127;
        async_copy_b128(lds_addr(&Ks[r][c]), src, 2 * e);
      }
    }
    {  // stage V tile transposed (needs VGPR round trip)
      const int r = tid >> 1, c0 = (tid & 1) * 64;
      const bf16_t* src = vb + base + (size_t)(kb0 * BKV + r) * HD + c0;
#pragma unroll
      for (int j = 0; j < 8; ++j) {
        union { u32x4 u; bf16_t e[8]; } t;
        t.u = *(const u32x4*)(src + j * 8);
#pragma unroll
        for (int e = 0; e < 8; ++e) Vt[c0 + j * 8 + e][r] = t.e[e];
      }
    }
    wait_asynccnt0();
    __syncthreads();

    if (kb0 * BKV <= myqmax) {  // wave-uniform: EXEC stays all-ones for WMMA
      // scores S = Q K^T : 4 key-subtiles x 4 k-steps
      v8f sacc[4] = {};
#pragma unroll
      for (int nt = 0; nt < 4; ++nt)
#pragma unroll
        for (int ksp = 0; ksp < 4; ++ksp) {
          v16bf bfr = load_b_frag(&Ks[nt * 16 + l16][ksp * 32], half);
          sacc[nt] = wmma_bf16(qf[ksp], bfr, sacc[nt]);
        }

      // scale + causal mask + row max
      float rmax[8];
#pragma unroll
      for (int r = 0; r < 8; ++r) rmax[r] = -1e30f;
#pragma unroll
      for (int nt = 0; nt < 4; ++nt) {
        const int kg = kb0 * BKV + nt * 16 + l16;
#pragma unroll
        for (int r = 0; r < 8; ++r) {
          float sv = sacc[nt][r] * ATTN_SCALE;
          const int qg = q0 + wave * 16 + r + half * 8;
          sv = (kg <= qg) ? sv : -1e30f;
          sacc[nt][r] = sv;
          rmax[r] = fmaxf(rmax[r], sv);
        }
      }
#pragma unroll
      for (int off = 1; off < 16; off <<= 1)
#pragma unroll
        for (int r = 0; r < 8; ++r)
          rmax[r] = fmaxf(rmax[r], __shfl_xor(rmax[r], off, 32));

      float corr[8];
#pragma unroll
      for (int r = 0; r < 8; ++r) {
        const float nm = fmaxf(mi[r], rmax[r]);
        corr[r] = __expf(mi[r] - nm);
        mi[r] = nm;
      }

      // P = exp(S - m), row sums, spill P to per-wave LDS (C->A relayout)
      float rsum[8] = {};
#pragma unroll
      for (int nt = 0; nt < 4; ++nt)
#pragma unroll
        for (int r = 0; r < 8; ++r) {
          const float p = __expf(sacc[nt][r] - mi[r]);
          rsum[r] += p;
          Ps[wave][r + half * 8][nt * 16 + l16] = (bf16_t)p;
        }
#pragma unroll
      for (int off = 1; off < 16; off <<= 1)
#pragma unroll
        for (int r = 0; r < 8; ++r) rsum[r] += __shfl_xor(rsum[r], off, 32);
#pragma unroll
      for (int r = 0; r < 8; ++r) li[r] = li[r] * corr[r] + rsum[r];
#pragma unroll
      for (int dt = 0; dt < 8; ++dt)
#pragma unroll
        for (int r = 0; r < 8; ++r) oacc[dt][r] *= corr[r];

      // O += P V : 2 key k-steps x 8 d-subtiles
#pragma unroll
      for (int ksp = 0; ksp < 2; ++ksp) {
        v16bf pf = load_a_frag(&Ps[wave][l16][ksp * 32], half);
#pragma unroll
        for (int dt = 0; dt < 8; ++dt) {
          v16bf vfr = load_b_frag(&Vt[dt * 16 + l16][ksp * 32], half);
          oacc[dt] = wmma_bf16(pf, vfr, oacc[dt]);
        }
      }
    }
  }

  // normalize and store ctx (bf16, [B,S,D] with d = h*HD + ...)
  const int b = bh / NH, h = bh % NH;
#pragma unroll
  for (int dt = 0; dt < 8; ++dt)
#pragma unroll
    for (int r = 0; r < 8; ++r) {
      const float o = oacc[dt][r] / li[r];
      const int srow = q0 + wave * 16 + r + half * 8;
      ctx[((size_t)(b * SEQ + srow)) * DIM + h * HD + dt * 16 + l16] = (bf16_t)o;
    }
}

// ---------------- host launcher ------------------------------------------------
extern "C" void kernel_launch(void* const* d_in, const int* in_sizes, int n_in,
                              void* d_out, int out_size, void* d_ws, size_t ws_size,
                              hipStream_t stream) {
  (void)in_sizes; (void)n_in; (void)out_size; (void)ws_size;
  const float* x    = (const float*)d_in[0];
  const float* Wqkv = (const float*)d_in[1];
  const float* bqkv = (const float*)d_in[2];
  const float* Wout = (const float*)d_in[3];
  const float* bout = (const float*)d_in[4];
  float* out = (float*)d_out;

  const size_t QKV_ELEMS = (size_t)BATCH * NH * SEQ * HD;  // 8,388,608
  char* ws = (char*)d_ws;
  bf16_t* qb  = (bf16_t*)(ws);
  bf16_t* kb  = (bf16_t*)(ws + QKV_ELEMS * 2);
  bf16_t* vb  = (bf16_t*)(ws + QKV_ELEMS * 4);
  bf16_t* ctx = (bf16_t*)(ws + QKV_ELEMS * 6);   // total 67.1 MB of workspace

  // 1) QKV projection: x(fp32) @ Wqkv(fp32) + bqkv -> q/k/v bf16 [B,H,S,HD]
  gemm_wmma_kernel<false, true>
      <<<dim3(3 * DIM / BN, M_ROWS / BM), 256, 0, stream>>>(
          x, nullptr, Wqkv, bqkv, qb, kb, vb, nullptr, 3 * DIM, DIM);

  // 2) RoPE on q and k
  {
    const int total = BATCH * NH * SEQ * 64;
    rope_kernel<<<total / 256, 256, 0, stream>>>(qb, kb);
  }

  // 3) causal flash attention -> ctx bf16 [B,S,D]
  flash_attn_kernel<<<dim3(BATCH * NH, SEQ / BQ), 128, 0, stream>>>(qb, kb, vb,
                                                                    ctx);

  // 4) output projection: ctx(bf16) @ Wout(fp32) + bout -> fp32 d_out
  gemm_wmma_kernel<true, false>
      <<<dim3(DIM / BN, M_ROWS / BM), 256, 0, stream>>>(
          nullptr, ctx, Wout, bout, nullptr, nullptr, nullptr, out, DIM, DIM);
}
